// MultiheadAttention_27358941676160
// MI455X (gfx1250) — matrix operand order repaired and checked by reading, not compiled, not run on hardware
//
#include <hip/hip_runtime.h>
#include <hip/hip_bf16.h>

// ---------------------------------------------------------------------------
// Causal MHA forward for MI455X (gfx1250), wave32 + WMMA bf16.
//   B=2, T=2048, D_IN=1024, H=16, D_HEAD=64, D_INNER=1024, D_OUT=64
// Pipeline: cvt(x,W->bf16, W transposed) -> QKV WMMA GEMM (double-buffered
// async global->LDS weight staging, ASYNCcnt-tracked) -> flash attention
// (online softmax, no materialized TxT scores) -> out-proj WMMA GEMM (+bias).
// ---------------------------------------------------------------------------

typedef __bf16 bf16_t;
typedef __attribute__((ext_vector_type(16))) __bf16 v16bf;
typedef __attribute__((ext_vector_type(8)))  __bf16 v8bf;
typedef __attribute__((ext_vector_type(8)))  float  v8f;
typedef __attribute__((ext_vector_type(4)))  int    v4i;

#define B_     2
#define T_     2048
#define DIN    1024
#define H_     16
#define DH     64
#define DINNER 1024
#define DOUT   64
#define NROW   (B_ * T_)   // 4096

#if __has_builtin(__builtin_amdgcn_global_load_async_to_lds_b128)
#define HAS_ASYNC_LDS 1
#else
#define HAS_ASYNC_LDS 0
#endif

typedef v4i __attribute__((address_space(1))) gm_v4i;
typedef v4i __attribute__((address_space(3))) lds_v4i;

// Async 16B global->LDS copy (ASYNCcnt-tracked); falls back to a synchronous
// VGPR round-trip with identical semantics under the same barrier schedule.
__device__ __forceinline__ void cp_b128_to_lds(bf16_t* lds, const bf16_t* g) {
#if HAS_ASYNC_LDS
  __builtin_amdgcn_global_load_async_to_lds_b128((gm_v4i*)g, (lds_v4i*)lds, 0, 0);
#else
  *(v8bf*)lds = *(const v8bf*)g;
#endif
}

template <int N>
__device__ __forceinline__ void wait_async() {
#if HAS_ASYNC_LDS
#if __has_builtin(__builtin_amdgcn_s_wait_asynccnt)
  __builtin_amdgcn_s_wait_asynccnt(N);
#else
  asm volatile("s_wait_asynccnt %0" ::"i"(N) : "memory");
#endif
#endif
}

__device__ __forceinline__ bf16_t f2bf(float f) {
  union { float f; unsigned u; } x; x.f = f;
  unsigned r = (x.u + 0x7FFFu + ((x.u >> 16) & 1u)) >> 16;   // RNE
  union { unsigned short s; bf16_t b; } y; y.s = (unsigned short)r;
  return y.b;
}

// A operand fragment: 16(M) x 32(K) bf16, from row-major [M][K] tile.
// ISA layout: lane = M (mod 16); lanes 0-15 hold K {0..7,16..23},
// lanes 16-31 hold K {8..15,24..31}; element i<8 -> lo, i>=8 -> hi.
__device__ __forceinline__ v16bf load_a16x32(const bf16_t* tile, int stride, int lane) {
  const int row = lane & 15;
  const int k0  = (lane & 16) ? 8 : 0;
  const bf16_t* p = tile + (size_t)row * stride + k0;
  v8bf lo = *(const v8bf*)(p);
  v8bf hi = *(const v8bf*)(p + 16);
  v16bf r;
#pragma unroll
  for (int i = 0; i < 8; ++i) { r[i] = lo[i]; r[i + 8] = hi[i]; }
  return r;
}

// B operand fragment: 32(K) x 16(N) bf16, sourced from B^T row-major [N][K].
// ISA layout: lane = N (mod 16); lanes 0-15 hold K {0..7,16..23},
// lanes 16-31 hold K {8..15,24..31}; element i<8 -> lo, i>=8 -> hi.
__device__ __forceinline__ v16bf load_bT32x16(const bf16_t* tileT, int stride, int lane) {
  const int row = lane & 15;              // N
  const int k0  = (lane & 16) ? 8 : 0;    // K base
  const bf16_t* p = tileT + (size_t)row * stride + k0;
  v8bf lo = *(const v8bf*)(p);
  v8bf hi = *(const v8bf*)(p + 16);
  v16bf r;
#pragma unroll
  for (int i = 0; i < 8; ++i) { r[i] = lo[i]; r[i + 8] = hi[i]; }
  return r;
}

__device__ __forceinline__ v8f wmma_bf16(v16bf a, v16bf b, v8f c) {
  // D(f32 16x16) = A(16x32 bf16) * B(32x16 bf16) + C
  return __builtin_amdgcn_wmma_f32_16x16x32_bf16(false, a, false, b,
                                                 (short)0, c, false, false);
}

// ------------------------------ converters --------------------------------
__global__ void k_cvt(const float* __restrict__ src, bf16_t* __restrict__ dst, int n) {
  int i = blockIdx.x * blockDim.x + threadIdx.x;
  if (i < n) dst[i] = f2bf(src[i]);
}

// src[rows][cols] (f32) -> dst[cols][rows] (bf16)
__global__ void k_transpose_cvt(const float* __restrict__ src, bf16_t* __restrict__ dst,
                                int rows, int cols) {
  int i = blockIdx.x * blockDim.x + threadIdx.x;
  if (i < rows * cols) {
    int r = i / cols, c = i % cols;
    dst[(size_t)c * rows + r] = f2bf(src[i]);
  }
}

// ------------------------- QKV projection GEMM ----------------------------
// C[4096 x 1024] = X(bf16) @ W ; B-fragments come from W^T rows.
// Weight tile (64x32) is double-buffered in LDS and filled with async
// global->LDS copies one k-step ahead of the WMMA consumer.
// mode 0 -> Q[b,h,t,d], mode 1 -> K[b,h,t,d], mode 2 -> V^T[b,h,d,t]
__global__ __launch_bounds__(128)
void k_qkv_gemm(const bf16_t* __restrict__ X, const bf16_t* __restrict__ WT,
                bf16_t* __restrict__ out, int mode) {
  __shared__ bf16_t Bs[2][64 * 32];
  const int lane = threadIdx.x & 31, wave = threadIdx.x >> 5;
  const int m0 = blockIdx.x * 64, n0 = blockIdx.y * 64;
  const int half = (lane >> 4) & 1;

  v8f acc[4];
#pragma unroll
  for (int nc = 0; nc < 4; ++nc)
#pragma unroll
    for (int i = 0; i < 8; ++i) acc[nc][i] = 0.0f;

  const bf16_t* arow = X + (size_t)(m0 + wave * 16) * DIN;

  // per-thread staging coordinates: 128 threads x 32B = 64x32 bf16 tile
  const int sr = threadIdx.x >> 1;          // 0..63 (W^T row within tile)
  const int sc = (threadIdx.x & 1) * 16;    // 0 or 16 (K column)
  const bf16_t* wrow = WT + (size_t)(n0 + sr) * DIN + sc;

  // prologue: stage tile 0 into buffer 0 (2 async b128 per thread)
  cp_b128_to_lds(&Bs[0][sr * 32 + sc],     wrow);
  cp_b128_to_lds(&Bs[0][sr * 32 + sc + 8], wrow + 8);

  int buf = 0;
  for (int k0 = 0; k0 < DIN; k0 += 32, buf ^= 1) {
    if (k0 + 32 < DIN) {
      // prefetch next tile into the other buffer (freed by last iteration's
      // trailing barrier)
      cp_b128_to_lds(&Bs[buf ^ 1][sr * 32 + sc],     wrow + (k0 + 32));
      cp_b128_to_lds(&Bs[buf ^ 1][sr * 32 + sc + 8], wrow + (k0 + 32) + 8);
      wait_async<2>();   // in-order: drains the 2 older (tile-k0) requests
    } else {
      wait_async<0>();
    }
    __syncthreads();     // tile k0 visible to all waves

    v16bf a = load_a16x32(arow + k0, DIN, lane);
#pragma unroll
    for (int nc = 0; nc < 4; ++nc) {
      v16bf b = load_bT32x16(&Bs[buf][nc * 16 * 32], 32, lane);
      acc[nc] = wmma_bf16(a, b, acc[nc]);
    }
    __syncthreads();     // protect Bs[buf] before it is refilled next+1 iter
  }

#pragma unroll
  for (int nc = 0; nc < 4; ++nc)
#pragma unroll
    for (int i = 0; i < 8; ++i) {
      int row = m0 + wave * 16 + i + half * 8;  // b*T + t
      int col = n0 + nc * 16 + (lane & 15);     // h*DH + d
      int b = row >> 11, t = row & (T_ - 1);
      int h = col >> 6,  d = col & (DH - 1);
      size_t idx;
      if (mode == 2) idx = ((size_t)(b * H_ + h) * DH + d) * T_ + t;  // V^T
      else           idx = ((size_t)(b * H_ + h) * T_ + t) * DH + d;  // Q / K
      out[idx] = f2bf(acc[nc][i]);
    }
}

// --------------------------- flash attention ------------------------------
// One WG = 4 waves; each wave owns 16 q-rows (WG covers 64 q-rows of one head).
__global__ __launch_bounds__(128)
void k_attn(const bf16_t* __restrict__ Q, const bf16_t* __restrict__ K,
            const bf16_t* __restrict__ Vt, bf16_t* __restrict__ Y) {
  __shared__ bf16_t Pbuf[4][16 * 32];   // wave-private C->A transpose buffers
  const int lane = threadIdx.x & 31, wave = threadIdx.x >> 5;
  const int bh = blockIdx.x;            // b*H + h
  const int qt = blockIdx.y;            // 64-row q tile
  const int half = (lane >> 4) & 1;

  const bf16_t* Qh = Q  + (size_t)bh * T_ * DH;
  const bf16_t* Kh = K  + (size_t)bh * T_ * DH;
  const bf16_t* Vh = Vt + (size_t)bh * DH * T_;
  const int q0 = qt * 64 + wave * 16;

  v16bf qa0 = load_a16x32(Qh + (size_t)q0 * DH,      DH, lane);
  v16bf qa1 = load_a16x32(Qh + (size_t)q0 * DH + 32, DH, lane);

  v8f yacc[4];
  float m_i[8], l_i[8];
#pragma unroll
  for (int i = 0; i < 8; ++i) { m_i[i] = -3.0e38f; l_i[i] = 0.0f; }
#pragma unroll
  for (int nc = 0; nc < 4; ++nc)
#pragma unroll
    for (int i = 0; i < 8; ++i) yacc[nc][i] = 0.0f;

  const float scale = 0.03125f;         // 1/sqrt(D_INNER) = 1/32
  bf16_t* pb = Pbuf[wave];
  const int kend = q0 + 16;             // causal: k <= q, wave-uniform bound

  for (int kt = 0; kt < kend; kt += 32) {
    // ---- S = Q K^T for 16(q) x 32(k); contraction over d (2 frags) ----
    v8f s0, s1;
#pragma unroll
    for (int i = 0; i < 8; ++i) { s0[i] = 0.0f; s1[i] = 0.0f; }
    {
      v16bf kb00 = load_bT32x16(Kh + (size_t)(kt)      * DH,      DH, lane);
      v16bf kb01 = load_bT32x16(Kh + (size_t)(kt)      * DH + 32, DH, lane);
      s0 = wmma_bf16(qa0, kb00, s0);
      s0 = wmma_bf16(qa1, kb01, s0);
      v16bf kb10 = load_bT32x16(Kh + (size_t)(kt + 16) * DH,      DH, lane);
      v16bf kb11 = load_bT32x16(Kh + (size_t)(kt + 16) * DH + 32, DH, lane);
      s1 = wmma_bf16(qa0, kb10, s1);
      s1 = wmma_bf16(qa1, kb11, s1);
    }
    // ---- scale + causal mask (selects, no divergence) ----
    float rmax[8];
#pragma unroll
    for (int i = 0; i < 8; ++i) {
      int qrow = q0 + i + half * 8;
      int kcol = kt + (lane & 15);
      float a0 = s0[i] * scale; if (kcol      > qrow) a0 = -3.0e38f;
      float a1 = s1[i] * scale; if (kcol + 16 > qrow) a1 = -3.0e38f;
      s0[i] = a0; s1[i] = a1;
      rmax[i] = fmaxf(a0, a1);
    }
    // row reductions across each 16-lane group (xor masks stay in-group)
#pragma unroll
    for (int off = 1; off < 16; off <<= 1)
#pragma unroll
      for (int i = 0; i < 8; ++i)
        rmax[i] = fmaxf(rmax[i], __shfl_xor(rmax[i], off, 32));

    float alpha[8], rsum[8];
#pragma unroll
    for (int i = 0; i < 8; ++i) {
      float mnew = fmaxf(m_i[i], rmax[i]);
      alpha[i] = __expf(m_i[i] - mnew);
      m_i[i] = mnew;
      float p0 = __expf(s0[i] - mnew);
      float p1 = __expf(s1[i] - mnew);
      s0[i] = p0; s1[i] = p1;
      rsum[i] = p0 + p1;
    }
#pragma unroll
    for (int off = 1; off < 16; off <<= 1)
#pragma unroll
      for (int i = 0; i < 8; ++i)
        rsum[i] += __shfl_xor(rsum[i], off, 32);
#pragma unroll
    for (int i = 0; i < 8; ++i) l_i[i] = l_i[i] * alpha[i] + rsum[i];
#pragma unroll
    for (int nc = 0; nc < 4; ++nc)
#pragma unroll
      for (int i = 0; i < 8; ++i) yacc[nc][i] *= alpha[i];

    // ---- P (C layout) -> LDS -> A fragment; wave-private, DS is in-order ----
#pragma unroll
    for (int i = 0; i < 8; ++i) {
      int r = i + half * 8;
      pb[r * 32 +      (lane & 15)] = f2bf(s0[i]);
      pb[r * 32 + 16 + (lane & 15)] = f2bf(s1[i]);
    }
    asm volatile("s_wait_dscnt 0" ::: "memory");
    v16bf pa = load_a16x32(pb, 32, lane);

    // ---- y += P @ V ; B-fragments from V^T rows (contiguous in t) ----
#pragma unroll
    for (int dc = 0; dc < 4; ++dc) {
      v16bf vb = load_bT32x16(Vh + (size_t)(dc * 16) * T_ + kt, T_, lane);
      yacc[dc] = wmma_bf16(pa, vb, yacc[dc]);
    }
  }

  // ---- finalize: divide by l, store to Y[(b*T+t)][h*DH+d] (bf16) ----
  const int b = bh >> 4, h = bh & 15;
#pragma unroll
  for (int i = 0; i < 8; ++i) {
    float inv = 1.0f / l_i[i];
    int t = qt * 64 + wave * 16 + i + half * 8;
    size_t rowoff = ((size_t)b * T_ + t) * DINNER + (size_t)h * DH;
#pragma unroll
    for (int dc = 0; dc < 4; ++dc)
      Y[rowoff + dc * 16 + (lane & 15)] = f2bf(yacc[dc][i] * inv);
  }
}

// ----------------------------- output proj --------------------------------
// out[4096 x 64] = Y(bf16)[4096 x 1024] @ Wo + bo, f32 output.
__global__ __launch_bounds__(128)
void k_proj(const bf16_t* __restrict__ Y, const bf16_t* __restrict__ WoT,
            const float* __restrict__ bo, float* __restrict__ out) {
  const int lane = threadIdx.x & 31, wave = threadIdx.x >> 5;
  const int m0 = blockIdx.x * 64;
  const int half = (lane >> 4) & 1;

  v8f acc[4];
#pragma unroll
  for (int nc = 0; nc < 4; ++nc)
#pragma unroll
    for (int i = 0; i < 8; ++i) acc[nc][i] = 0.0f;

  const bf16_t* arow = Y + (size_t)(m0 + wave * 16) * DINNER;
  for (int k0 = 0; k0 < DINNER; k0 += 32) {
    v16bf a = load_a16x32(arow + k0, DINNER, lane);
#pragma unroll
    for (int nc = 0; nc < 4; ++nc) {
      v16bf b = load_bT32x16(WoT + (size_t)(nc * 16) * DINNER + k0, DINNER, lane);
      acc[nc] = wmma_bf16(a, b, acc[nc]);
    }
  }
#pragma unroll
  for (int nc = 0; nc < 4; ++nc)
#pragma unroll
    for (int i = 0; i < 8; ++i) {
      int row = m0 + wave * 16 + i + half * 8;
      int col = nc * 16 + (lane & 15);
      out[(size_t)row * DOUT + col] = acc[nc][i] + bo[col];
    }
}

// ------------------------------ launcher ----------------------------------
extern "C" void kernel_launch(void* const* d_in, const int* in_sizes, int n_in,
                              void* d_out, int out_size, void* d_ws, size_t ws_size,
                              hipStream_t stream) {
  (void)in_sizes; (void)n_in; (void)out_size; (void)ws_size;
  const float* x  = (const float*)d_in[0];
  const float* Wq = (const float*)d_in[1];
  const float* Wk = (const float*)d_in[2];
  const float* Wv = (const float*)d_in[3];
  const float* Wo = (const float*)d_in[4];
  const float* bo = (const float*)d_in[5];
  float* out = (float*)d_out;

  // workspace layout (bytes), total ~46.1 MB
  char* ws = (char*)d_ws;
  bf16_t* xb  = (bf16_t*)(ws);              // 4096x1024        ( 8 MB)
  bf16_t* WqT = (bf16_t*)(ws +  8388608);   // 1024x1024 W^T    ( 2 MB)
  bf16_t* WkT = (bf16_t*)(ws + 10485760);
  bf16_t* WvT = (bf16_t*)(ws + 12582912);
  bf16_t* WoT = (bf16_t*)(ws + 14680064);   //   64x1024 Wo^T
  bf16_t* Qb  = (bf16_t*)(ws + 14811136);   // [b,h,t,d]        ( 8 MB)
  bf16_t* Kb  = (bf16_t*)(ws + 23199744);   // [b,h,t,d]        ( 8 MB)
  bf16_t* Vtb = (bf16_t*)(ws + 31588352);   // [b,h,d,t]        ( 8 MB)
  bf16_t* Yat = (bf16_t*)(ws + 39976960);   // [b*t, h*d]       ( 8 MB)

  k_cvt<<<(NROW * DIN) / 256, 256, 0, stream>>>(x, xb, NROW * DIN);
  k_transpose_cvt<<<(DIN * DINNER) / 256, 256, 0, stream>>>(Wq, WqT, DIN, DINNER);
  k_transpose_cvt<<<(DIN * DINNER) / 256, 256, 0, stream>>>(Wk, WkT, DIN, DINNER);
  k_transpose_cvt<<<(DIN * DINNER) / 256, 256, 0, stream>>>(Wv, WvT, DIN, DINNER);
  k_transpose_cvt<<<(DINNER * DOUT) / 256, 256, 0, stream>>>(Wo, WoT, DINNER, DOUT);

  dim3 gq(NROW / 64, DINNER / 64);
  k_qkv_gemm<<<gq, 128, 0, stream>>>(xb, WqT, Qb, 0);
  k_qkv_gemm<<<gq, 128, 0, stream>>>(xb, WkT, Kb, 1);
  k_qkv_gemm<<<gq, 128, 0, stream>>>(xb, WvT, Vtb, 2);

  k_attn<<<dim3(B_ * H_, T_ / 64), 128, 0, stream>>>(Qb, Kb, Vtb, Yat);
  k_proj<<<NROW / 64, 128, 0, stream>>>(Yat, WoT, bo, out);
}
